// MultiHeadAttention_40716289966109
// MI455X (gfx1250) — matrix operand order repaired and checked by reading, not compiled, not run on hardware
//
#include <hip/hip_runtime.h>

// ---------------------------------------------------------------------------
// MultiHeadAttention (B=4, N=2048, E=512, H=8, D=64), fp32 in/out.
// fp32 -> bf16 on the fly, v_wmma_f32_16x16x32_bf16 everywhere, fp32
// accumulate + fp32 online softmax. Q is pre-scaled by (1/sqrt(E))*log2(e)
// at scatter time so attention uses exp2 (v_exp_f32 is base-2 natively) and
// needs no per-step scale muls. Softmax row-sums are computed by an extra
// WMMA against an all-ones B tile (C-layout broadcasts the row value across
// the 16-lane half -- exactly the pattern the online update needs), removing
// half of the ds_bpermute butterflies. All LDS tiles use 40-element (80 B)
// row stride: 16B-aligned rows (ds_load_b128) and bank-conflict free.
// ---------------------------------------------------------------------------

#define B_   4
#define N_   2048
#define E_   512
#define H_   8
#define D_   64
#define E3_  1536
#define LSTR 40   // LDS row stride in bf16 elements

typedef __bf16 bf16;
typedef __attribute__((ext_vector_type(16))) __bf16 v16bf;
typedef __attribute__((ext_vector_type(8)))  __bf16 v8bf;
typedef __attribute__((ext_vector_type(4)))  __bf16 v4bf;
typedef __attribute__((ext_vector_type(8)))  float  v8f;

static __device__ __forceinline__ v8f wmma_bf16(v16bf a, v16bf b, v8f c) {
  // (neg_a, A, neg_b, B, c_mod, C, reuse_a, reuse_b)
  return __builtin_amdgcn_wmma_f32_16x16x32_bf16(false, a, false, b,
                                                 (short)0, c, false, false);
}

static __device__ __forceinline__ v16bf mk16(v8bf lo, v8bf hi) {
  return __builtin_shufflevector(lo, hi, 0, 1, 2, 3, 4, 5, 6, 7,
                                 8, 9, 10, 11, 12, 13, 14, 15);
}

static __device__ __forceinline__ v8f v8f_zero() {
  v8f z = {0.f, 0.f, 0.f, 0.f, 0.f, 0.f, 0.f, 0.f};
  return z;
}

// ---------------------------------------------------------------------------
// Tiled GEMM: C[M,Nld] = A[M,K] * Bm[K,Nld] + bias.
// Block = 256 threads = 8 waves; block tile 64(M) x 64(N); K-step 32.
// A staged row-major [64][40]; B staged TRANSPOSED [n][k] so both fragment
// types are contiguous 16B chunks (ds_load_b128).
// ---------------------------------------------------------------------------
template <typename AT, bool QKV_SCATTER>
__global__ __launch_bounds__(256) void gemm_bf16_wmma(
    const AT* __restrict__ A, const float* __restrict__ Bm,
    const float* __restrict__ bias, float* __restrict__ Out,
    bf16* __restrict__ qkv_ws, int M, int K, int Nld) {
  __shared__ __align__(16) bf16 As[64][LSTR];
  __shared__ __align__(16) bf16 Bt[64][LSTR];  // [n][k]

  const int tid  = threadIdx.x;
  const int wave = tid >> 5;
  const int lane = tid & 31;
  const int half = lane >> 4;
  const int lr   = lane & 15;

  const int m0 = blockIdx.y * 64;
  const int n0 = blockIdx.x * 64;

  const int tr    = wave & 3;   // 0..3: 16-row strip
  const int tc    = wave >> 2;  // 0..1: 32-col strip
  const int mrow  = tr * 16;
  const int ncol0 = tc * 32;

  v8f acc[2];
  acc[0] = v8f_zero();
  acc[1] = v8f_zero();

  for (int k0 = 0; k0 < K; k0 += 32) {
    // Stage A tile 64x32: 4 consecutive elems per thread (b128/b64 loads).
#pragma unroll
    for (int i = 0; i < 2; ++i) {
      const int vid = i * 256 + tid;
      const int r = vid >> 3, c = (vid & 7) * 4;
      const AT* ap = &A[(size_t)(m0 + r) * K + (k0 + c)];
      v4bf w;
#pragma unroll
      for (int j = 0; j < 4; ++j) w[j] = (bf16)ap[j];
      *(v4bf*)&As[r][c] = w;
    }
    // Stage B tile 32x64 transposed: coalesced b128 reads, scattered b16
    // writes (store conflicts are cheap; loads are the hot path).
#pragma unroll
    for (int i = 0; i < 2; ++i) {
      const int vid = i * 256 + tid;
      const int k = vid >> 4, c = (vid & 15) * 4;
      const float* bp = &Bm[(size_t)(k0 + k) * Nld + (n0 + c)];
#pragma unroll
      for (int j = 0; j < 4; ++j) Bt[c + j][k] = (bf16)bp[j];
    }
    __syncthreads();

    // A frag: row mrow+lr, K chunks {half*8.., half*8+16..} -> v8bf half, half+2
    const v8bf* arow = (const v8bf*)&As[mrow + lr][0];
    const v16bf a = mk16(arow[half], arow[half + 2]);
    // B frag: col = n, K chunks {half*8.., half*8+16..} -> v8bf half, half+2
    const v8bf* brow0 = (const v8bf*)&Bt[ncol0 + lr][0];
    const v8bf* brow1 = (const v8bf*)&Bt[ncol0 + 16 + lr][0];
    const v16bf b0 = mk16(brow0[half], brow0[half + 2]);
    const v16bf b1 = mk16(brow1[half], brow1[half + 2]);

    acc[0] = wmma_bf16(a, b0, acc[0]);
    acc[1] = wmma_bf16(a, b1, acc[1]);
    __syncthreads();
  }

  // Epilogue. C layout: VGPR r -> row = r + 8*half, col = lr.
  const int rowbase = m0 + mrow + half * 8;
#pragma unroll
  for (int t = 0; t < 2; ++t) {
    const int c  = n0 + ncol0 + t * 16 + lr;
    const float bv = bias[c];
    if constexpr (QKV_SCATTER) {
      // col c of [*,3E] splits (h, d, qkv): which=c%3, h=c/192, d=(c%192)/3
      const int which = c % 3;
      const int h     = c / (3 * D_);
      const int d     = (c % (3 * D_)) / 3;
      // Fold softmax scale (1/sqrt(E)) and log2(e) into Q so the attention
      // kernel uses exp2 with no per-step scaling.
      const float qscale =
          (float)(0.04419417382415922 * 1.4426950408889634);
      const float s = (which == 0) ? qscale : 1.f;
      bf16* dst = qkv_ws + (size_t)which * ((size_t)B_ * H_ * N_ * D_);
#pragma unroll
      for (int r = 0; r < 8; ++r) {
        const int grow = rowbase + r;  // global row in [0, B*N)
        const int bb = grow / N_, nn = grow % N_;
        dst[((((size_t)bb * H_) + h) * N_ + nn) * D_ + d] =
            (bf16)((acc[t][r] + bv) * s);
      }
    } else {
#pragma unroll
      for (int r = 0; r < 8; ++r) {
        Out[(size_t)(rowbase + r) * Nld + c] = acc[t][r] + bv;
      }
    }
  }
}

// ---------------------------------------------------------------------------
// Flash attention. Q,K,V bf16 [B,H,N,D]; Q pre-scaled (see above).
// Block = 8 waves = 128 query rows of one (b,h); each wave owns 16 rows.
// Per 32-key step: 4 WMMAs for S (log2-domain scores), max via 16-lane
// butterflies, P -> LDS transpose, then 4 WMMAs for PV plus 1 WMMA against
// an all-ones tile for the softmax row sums.
// ---------------------------------------------------------------------------
__global__ __launch_bounds__(256) void flash_attn_wmma(
    const bf16* __restrict__ Q, const bf16* __restrict__ Kx,
    const bf16* __restrict__ V, bf16* __restrict__ Obuf) {
  __shared__ __align__(16) bf16 Pl[8][16][LSTR];  // per-wave P (16x32)
  __shared__ __align__(16) bf16 Vt[64][LSTR];     // V tile transposed [d][k]

  const int tid  = threadIdx.x;
  const int wave = tid >> 5;
  const int lane = tid & 31;
  const int half = lane >> 4;
  const int lr   = lane & 15;

  const int bh = blockIdx.y;  // 0..B*H-1
  const int b  = bh >> 3;     // /H
  const int h  = bh & 7;      // %H
  const int q0 = blockIdx.x * 128 + wave * 16;

  const bf16* Qp = Q + ((size_t)bh * N_ + q0) * D_;
  const bf16* Kp = Kx + (size_t)bh * N_ * D_;
  const bf16* Vp = V + (size_t)bh * N_ * D_;

  // Q A-fragments for both K-steps (D=64 -> 2 x 32), kept in registers.
  const v8bf* qrow = (const v8bf*)(Qp + (size_t)lr * D_);
  const v16bf aq0 = mk16(qrow[half], qrow[half + 2]);
  const v16bf aq1 = mk16(qrow[half + 4], qrow[half + 6]);

  // All-ones B fragment: P x ones = row sums of P, broadcast per C layout.
  v16bf ones;
#pragma unroll
  for (int j = 0; j < 16; ++j) ones[j] = (bf16)1.0f;

  float mrun[8];
  v8f accO[4], accS;
#pragma unroll
  for (int r = 0; r < 8; ++r) mrun[r] = -1e30f;
#pragma unroll
  for (int t = 0; t < 4; ++t) accO[t] = v8f_zero();
  accS = v8f_zero();

  for (int kb = 0; kb < N_; kb += 32) {
    // Cooperative stage of V tile 32(k) x 64(d), transposed: [d][k].
#pragma unroll
    for (int i = 0; i < 2; ++i) {
      const int vid = i * 256 + tid;
      const int k = vid >> 4, d4 = (vid & 15) * 4;
      const bf16* vp = Vp + (size_t)(kb + k) * D_ + d4;
#pragma unroll
      for (int j = 0; j < 4; ++j) Vt[d4 + j][k] = vp[j];
    }

    // K^T B-frags: col = key, rows = d. Contiguous global b128 reads.
    const v8bf* kr0 = (const v8bf*)(Kp + (size_t)(kb + lr) * D_);
    const v8bf* kr1 = (const v8bf*)(Kp + (size_t)(kb + 16 + lr) * D_);
    const v16bf bk00 = mk16(kr0[half], kr0[half + 2]);
    const v16bf bk01 = mk16(kr0[4 + half], kr0[6 + half]);
    const v16bf bk10 = mk16(kr1[half], kr1[half + 2]);
    const v16bf bk11 = mk16(kr1[4 + half], kr1[6 + half]);

    v8f s0 = v8f_zero(), s1 = v8f_zero();
    s0 = wmma_bf16(aq0, bk00, s0);
    s0 = wmma_bf16(aq1, bk01, s0);
    s1 = wmma_bf16(aq0, bk10, s1);
    s1 = wmma_bf16(aq1, bk11, s1);

    // Online softmax in log2 domain (Q carries scale*log2e). Butterfly
    // masks 1..8 stay inside each 16-lane half; row of VGPR r is fixed.
    float p0[8], p1[8], alpha[8];
#pragma unroll
    for (int r = 0; r < 8; ++r) {
      float mx = fmaxf(s0[r], s1[r]);
#pragma unroll
      for (int off = 8; off >= 1; off >>= 1)
        mx = fmaxf(mx, __shfl_xor(mx, off, 32));
      const float mnew = fmaxf(mrun[r], mx);
      p0[r]    = exp2f(s0[r] - mnew);
      p1[r]    = exp2f(s1[r] - mnew);
      alpha[r] = exp2f(mrun[r] - mnew);
      mrun[r]  = mnew;
    }
#pragma unroll
    for (int t = 0; t < 4; ++t)
#pragma unroll
      for (int r = 0; r < 8; ++r) accO[t][r] *= alpha[r];
#pragma unroll
    for (int r = 0; r < 8; ++r) accS[r] *= alpha[r];

    // P: C layout -> A layout through per-wave LDS.
#pragma unroll
    for (int r = 0; r < 8; ++r) {
      Pl[wave][half * 8 + r][lr]      = (bf16)p0[r];
      Pl[wave][half * 8 + r][16 + lr] = (bf16)p1[r];
    }
    __syncthreads();  // publishes Vt + Pl; uniform trip count
    const v8bf* prow = (const v8bf*)&Pl[wave][lr][0];
    const v16bf pa = mk16(prow[half], prow[half + 2]);

    // PV: O[16x16 per d-tile] += P(16x32) x V(32x16), 4 d-tiles, b128 frags.
#pragma unroll
    for (int t2 = 0; t2 < 4; ++t2) {
      const v8bf* vrow0 = (const v8bf*)&Vt[t2 * 16 + lr][0];
      const v16bf bv = mk16(vrow0[half], vrow0[half + 2]);
      accO[t2] = wmma_bf16(pa, bv, accO[t2]);
    }
    // Row sums of P via WMMA (replaces the sum butterflies).
    accS = wmma_bf16(pa, ones, accS);
    __syncthreads();  // protect Vt/Pl against next iteration's writes

    if (kb + 32 < N_) {
      __builtin_prefetch(Kp + (size_t)(kb + 32 + lr) * D_, 0, 1);
      __builtin_prefetch(Vp + (size_t)(kb + 32 + lr) * D_, 0, 1);
    }
  }

  // Normalize and write O as bf16 [B, N, E] (col = h*64 + d).
  float inv[8];
#pragma unroll
  for (int r = 0; r < 8; ++r) inv[r] = 1.f / accS[r];
  bf16* Op = Obuf + (size_t)b * N_ * E_;
#pragma unroll
  for (int t2 = 0; t2 < 4; ++t2)
#pragma unroll
    for (int r = 0; r < 8; ++r) {
      const int n = q0 + half * 8 + r;
      Op[(size_t)n * E_ + h * D_ + t2 * 16 + lr] = (bf16)(accO[t2][r] * inv[r]);
    }
}

// ---------------------------------------------------------------------------
extern "C" void kernel_launch(void* const* d_in, const int* in_sizes, int n_in,
                              void* d_out, int out_size, void* d_ws,
                              size_t ws_size, hipStream_t stream) {
  (void)in_sizes; (void)n_in; (void)out_size; (void)ws_size;
  const float* x      = (const float*)d_in[0];
  const float* W_qkv  = (const float*)d_in[1];
  const float* b_qkv  = (const float*)d_in[2];
  const float* W_proj = (const float*)d_in[3];
  const float* b_proj = (const float*)d_in[4];
  float* out = (float*)d_out;

  const size_t BHND = (size_t)B_ * H_ * N_ * D_;  // 4,194,304 elems
  bf16* qkv = (bf16*)d_ws;                        // Q,K,V bf16: 3*BHND
  bf16* Ob  = qkv + 3 * BHND;                     // O bf16: B*N*E
  // total workspace ~34 MB

  dim3 blk(256);
  // 1) x[8192,512] @ W_qkv[512,1536] + b -> scatter Q/K/V (Q pre-scaled)
  gemm_bf16_wmma<float, true>
      <<<dim3(E3_ / 64, (B_ * N_) / 64), blk, 0, stream>>>(
          x, W_qkv, b_qkv, nullptr, qkv, B_ * N_, E_, E3_);
  // 2) attention per (b,h), 128 query rows per block
  flash_attn_wmma<<<dim3(N_ / 128, B_ * H_), blk, 0, stream>>>(
      qkv, qkv + BHND, qkv + 2 * BHND, Ob);
  // 3) O[8192,512] @ W_proj[512,512] + b -> d_out (fp32)
  gemm_bf16_wmma<bf16, false>
      <<<dim3(E_ / 64, (B_ * N_) / 64), blk, 0, stream>>>(
          Ob, W_proj, b_proj, out, nullptr, B_ * N_, E_, E_);
}
